// GAIN_GLOVE_19851338842924
// MI455X (gfx1250) — compile-verified, matrix operand order checked by reading, not run on hardware
//
#include <hip/hip_runtime.h>
#include <hip/hip_bf16.h>

typedef __attribute__((ext_vector_type(16))) _Float16 v16h;
typedef __attribute__((ext_vector_type(8)))  float    v8f;

#define TPB 256
#define WPB 8   // waves per block (wave32)

// ---------------------------------------------------------------------------
// Generic WMMA GEMM.
//   NT=1 :  C[M,N] = A[M,K] * B[N,K]^T     (both row-major, k contiguous)
//   NT=0 :  C[M,N] = A[M,K] * B[K,N]       (B row-major [K,N])
// flags: 1=add bias[col], 2=relu, 4=accumulate into existing C
// aridx: optional indirect row index for A (row m -> aridx[m*aridx_stride])
// Each wave computes a 16(M) x 64(N) tile with 4 accumulators sharing the A
// fragment; f32 -> f16 conversion on load, f32 accumulate.
// ---------------------------------------------------------------------------
template <int NT>
__global__ void gemm_kernel(const float* __restrict__ A, int lda, long long sA,
                            const float* __restrict__ Bm, int ldb, long long sB,
                            float* __restrict__ C, int ldc, long long sC,
                            const float* __restrict__ bias,
                            int M, int N, int K, int flags,
                            const int* __restrict__ aridx, int aridx_stride)
{
    int tilesN = (N + 63) >> 6;
    int tilesM = (M + 15) >> 4;
    int wid = blockIdx.x * WPB + (threadIdx.x >> 5);
    if (wid >= tilesM * tilesN) return;      // uniform per wave
    int mt = wid / tilesN;
    int nt = wid % tilesN;

    int lane = threadIdx.x & 31;
    int ml = lane & 15;
    int kh = lane >> 4;

    A  += (long long)blockIdx.z * sA;
    Bm += (long long)blockIdx.z * sB;
    C  += (long long)blockIdx.z * sC;

    int row = mt * 16 + ml;
    long long arow = -1;
    if (row < M) arow = aridx ? (long long)aridx[(long long)row * aridx_stride]
                              : (long long)row;

    v8f acc[4] = {};

    for (int kb = 0; kb < K; kb += 32) {
        int k0 = kb + kh * 8;
        v16h a;
#pragma unroll
        for (int i = 0; i < 8; ++i) {
            int k  = k0 + i;
            int k2 = k + 16;
            a[i]     = (arow >= 0 && k  < K) ? (_Float16)A[arow * lda + k ] : (_Float16)0.f;
            a[i + 8] = (arow >= 0 && k2 < K) ? (_Float16)A[arow * lda + k2] : (_Float16)0.f;
        }
#pragma unroll
        for (int j = 0; j < 4; ++j) {
            int col = nt * 64 + j * 16 + ml;
            v16h b;
#pragma unroll
            for (int i = 0; i < 8; ++i) {
                int k  = k0 + i;
                int k2 = k + 16;
                if (NT) {
                    b[i]     = (col < N && k  < K) ? (_Float16)Bm[(long long)col * ldb + k ] : (_Float16)0.f;
                    b[i + 8] = (col < N && k2 < K) ? (_Float16)Bm[(long long)col * ldb + k2] : (_Float16)0.f;
                } else {
                    b[i]     = (col < N && k  < K) ? (_Float16)Bm[(long long)k  * ldb + col] : (_Float16)0.f;
                    b[i + 8] = (col < N && k2 < K) ? (_Float16)Bm[(long long)k2 * ldb + col] : (_Float16)0.f;
                }
            }
            acc[j] = __builtin_amdgcn_wmma_f32_16x16x32_f16(
                false, a, false, b, (short)0, acc[j], false, false);
        }
    }

#pragma unroll
    for (int j = 0; j < 4; ++j) {
        int col = nt * 64 + j * 16 + ml;
        if (col >= N) continue;
#pragma unroll
        for (int r = 0; r < 8; ++r) {
            int m = mt * 16 + r + kh * 8;
            if (m >= M) continue;
            long long o = (long long)m * ldc + col;
            float v = acc[j][r];
            if (flags & 4) v += C[o];
            if (flags & 1) v += bias[col];
            if (flags & 2) v = fmaxf(v, 0.f);
            C[o] = v;
        }
    }
}

// ---------------------------------------------------------------------------
// Fused BiLSTM encoder. One block per direction (blockIdx.x: 0=fwd, 1=bwd).
// State h,c [16,128] and gates [16,512] in LDS.  Per step: WMMA 16x512x128
// recurrence (exact tiles, EXEC all-ones), then gate nonlinearity, masked
// store of h into features[:, t, dir*128 : dir*128+128].
// ---------------------------------------------------------------------------
__global__ void enc_lstm_kernel(const float* __restrict__ GxF,
                                const float* __restrict__ GxB,
                                const float* __restrict__ WhhF,
                                const float* __restrict__ WhhB,
                                const float* __restrict__ bF,
                                const float* __restrict__ bB,
                                const float* __restrict__ mask,
                                float* __restrict__ features)
{
    __shared__ float s_h[16 * 128];
    __shared__ float s_c[16 * 128];
    __shared__ float s_g[16 * 512];

    int dir = blockIdx.x;
    const float* Gx   = dir ? GxB  : GxF;
    const float* Whh  = dir ? WhhB : WhhF;
    const float* bias = dir ? bB   : bF;

    int tid = threadIdx.x;
    for (int i = tid; i < 16 * 128; i += TPB) { s_h[i] = 0.f; s_c[i] = 0.f; }
    __syncthreads();

    int w = tid >> 5, lane = tid & 31, ml = lane & 15, kh = lane >> 4;

    for (int step = 0; step < 512; ++step) {
        int t = dir ? (511 - step) : step;

        v8f acc[4] = {};
        for (int kb = 0; kb < 128; kb += 32) {
            int k0 = kb + kh * 8;
            v16h a;
#pragma unroll
            for (int i = 0; i < 8; ++i) {
                a[i]     = (_Float16)s_h[ml * 128 + k0 + i];
                a[i + 8] = (_Float16)s_h[ml * 128 + k0 + 16 + i];
            }
#pragma unroll
            for (int j = 0; j < 4; ++j) {
                int col = (w * 4 + j) * 16 + ml;
                v16h b;
#pragma unroll
                for (int i = 0; i < 8; ++i) {
                    b[i]     = (_Float16)Whh[col * 128 + k0 + i];
                    b[i + 8] = (_Float16)Whh[col * 128 + k0 + 16 + i];
                }
                acc[j] = __builtin_amdgcn_wmma_f32_16x16x32_f16(
                    false, a, false, b, (short)0, acc[j], false, false);
            }
        }
#pragma unroll
        for (int j = 0; j < 4; ++j) {
            int col = (w * 4 + j) * 16 + ml;
#pragma unroll
            for (int r = 0; r < 8; ++r) {
                int m = r + kh * 8;    // batch row
                s_g[m * 512 + col] =
                    acc[j][r] + Gx[((long long)m * 512 + t) * 512 + col] + bias[col];
            }
        }
        __syncthreads();

        for (int i = tid; i < 16 * 128; i += TPB) {
            int m = i >> 7, j = i & 127;
            const float* g = s_g + m * 512;
            float ig = 1.f / (1.f + __expf(-g[j]));
            float fg = 1.f / (1.f + __expf(-g[128 + j]));
            float gg = tanhf(g[256 + j]);
            float og = 1.f / (1.f + __expf(-g[384 + j]));
            float c = fg * s_c[i] + ig * gg;
            float h = og * tanhf(c);
            s_c[i] = c; s_h[i] = h;
            features[((long long)m * 576 + t) * 1024 + dir * 128 + j] =
                h * mask[m * 512 + t];
        }
        __syncthreads();
    }
}

// ---------------------------------------------------------------------------
// Elementwise / small kernels
// ---------------------------------------------------------------------------
__global__ void zero_kernel(float* p, long long n) {
    long long i = (long long)blockIdx.x * blockDim.x + threadIdx.x;
    if (i < n) p[i] = 0.f;
}

__global__ void emb_kernel(const int* __restrict__ words, const int* __restrict__ etype,
                           const int* __restrict__ eid,
                           const float* __restrict__ wemb, const float* __restrict__ temb,
                           const float* __restrict__ iemb, float* __restrict__ emb)
{
    int i = blockIdx.x * blockDim.x + threadIdx.x;
    if (i >= 16 * 512) return;
    float* dst = emb + (long long)i * 140;
    const float* w = wemb + (long long)words[i] * 100;
    for (int j = 0; j < 100; ++j) dst[j] = w[j];
    const float* t = temb + (long long)etype[i] * 20;
    for (int j = 0; j < 20; ++j) dst[100 + j] = t[j];
    const float* d = iemb + (long long)eid[i] * 20;
    for (int j = 0; j < 20; ++j) dst[120 + j] = d[j];
}

__global__ void adjnorm_kernel(const float* __restrict__ adj, float* __restrict__ An) {
    int i = blockIdx.x * blockDim.x + threadIdx.x;  // b*576+row
    if (i >= 16 * 576) return;
    const float* r = adj + (long long)i * 576;
    float s = 0.f;
    for (int j = 0; j < 576; ++j) s += r[j];
    float inv = 1.f / (s + 1e-8f);
    float* o = An + (long long)i * 576;
    for (int j = 0; j < 576; ++j) o[j] = r[j] * inv;
}

__global__ void softmax_rows_kernel(float* __restrict__ X, long long rows, int cols,
                                    float scale)
{
    long long i = (long long)blockIdx.x * blockDim.x + threadIdx.x;
    if (i >= rows) return;
    float* r = X + i * cols;
    float mx = -3.4e38f;
    for (int j = 0; j < cols; ++j) { float v = r[j] * scale; r[j] = v; mx = fmaxf(mx, v); }
    float s = 0.f;
    for (int j = 0; j < cols; ++j) { float e = __expf(r[j] - mx); r[j] = e; s += e; }
    float inv = 1.f / s;
    for (int j = 0; j < cols; ++j) r[j] *= inv;
}

__global__ void add_into_features_kernel(float* __restrict__ features, int colofs,
                                         const float* __restrict__ a,
                                         const float* __restrict__ b)
{
    long long i = (long long)blockIdx.x * blockDim.x + threadIdx.x; // 9216*256
    if (i >= 9216LL * 256) return;
    long long row = i >> 8; int c = (int)(i & 255);
    features[row * 1024 + colofs + c] = a[i] + b[i];
}

__global__ void pair_gather_kernel(const float* __restrict__ features,
                                   const int* __restrict__ pairs,
                                   float* __restrict__ hf, float* __restrict__ tf)
{
    long long i = (long long)blockIdx.x * blockDim.x + threadIdx.x; // 512*1024
    if (i >= 512LL * 1024) return;
    int bp = (int)(i >> 10), d = (int)(i & 1023);
    int b = bp >> 5;
    int hi = pairs[bp * 2 + 0]; hi += (hi == 0); hi -= 1;
    int ti = pairs[bp * 2 + 1]; ti += (ti == 0); ti -= 1;
    hf[i] = features[(long long)(b * 576 + 512 + hi) * 1024 + d];
    tf[i] = features[(long long)(b * 576 + 512 + ti) * 1024 + d];
}

__global__ void gidx_kernel(const int* __restrict__ path_idx, int* __restrict__ gidx) {
    int i = blockIdx.x * blockDim.x + threadIdx.x;
    if (i >= 16384) return;
    int b = i >> 10;                 // P*K*L = 1024 per batch
    gidx[i] = b * 576 + path_idx[i];
}

__global__ void sub_kernel(const float* __restrict__ a, const float* __restrict__ b,
                           float* __restrict__ o, long long n)
{
    long long i = (long long)blockIdx.x * blockDim.x + threadIdx.x;
    if (i < n) o[i] = a[i] - b[i];
}

// LSTM gate update; optional running-max into pe (path encoding).
__global__ void lstm_gate_kernel(const float* __restrict__ G, const float* __restrict__ bias,
                                 float* __restrict__ c, float* __restrict__ h,
                                 float* __restrict__ pe, int first, int Bc, int H)
{
    long long i = (long long)blockIdx.x * blockDim.x + threadIdx.x;
    if (i >= (long long)Bc * H) return;
    int m = (int)(i / H), j = (int)(i % H);
    const float* g = G + (long long)m * 4 * H;
    float ig = 1.f / (1.f + __expf(-(g[j] + bias[j])));
    float fg = 1.f / (1.f + __expf(-(g[H + j] + bias[H + j])));
    float gg = tanhf(g[2 * H + j] + bias[2 * H + j]);
    float og = 1.f / (1.f + __expf(-(g[3 * H + j] + bias[3 * H + j])));
    float cc = fg * c[i] + ig * gg;
    float hh = og * tanhf(cc);
    c[i] = cc; h[i] = hh;
    if (pe) pe[i] = first ? hh : fmaxf(pe[i], hh);
}

// Multi-head (4 heads x 256) attention over K=4 paths, per (b,p,head) thread.
__global__ void pathattn_kernel(const float* __restrict__ Q, const float* __restrict__ Kp,
                                const float* __restrict__ Vp, float* __restrict__ O)
{
    int i = blockIdx.x * blockDim.x + threadIdx.x;  // 512 bp * 4 heads
    if (i >= 2048) return;
    int bp = i >> 2, h = i & 3;
    const float* q = Q + (long long)bp * 1024 + h * 256;
    float sc[4];
    for (int k = 0; k < 4; ++k) {
        const float* kk = Kp + ((long long)bp * 4 + k) * 1024 + h * 256;
        float s = 0.f;
        for (int d = 0; d < 256; ++d) s += q[d] * kk[d];
        sc[k] = s * (1.f / 16.f);
    }
    float mx = fmaxf(fmaxf(sc[0], sc[1]), fmaxf(sc[2], sc[3]));
    float ss = 0.f;
    for (int k = 0; k < 4; ++k) { sc[k] = __expf(sc[k] - mx); ss += sc[k]; }
    float inv = 1.f / ss;
    for (int k = 0; k < 4; ++k) sc[k] *= inv;
    float* o = O + (long long)bp * 1024 + h * 256;
    for (int d = 0; d < 256; ++d) {
        float s = 0.f;
        for (int k = 0; k < 4; ++k)
            s += sc[k] * Vp[((long long)bp * 4 + k) * 1024 + h * 256 + d];
        o[d] = s;
    }
}

__global__ void featcat_kernel(const float* __restrict__ hf, const float* __restrict__ tf,
                               const float* __restrict__ pi, float* __restrict__ feat)
{
    long long i = (long long)blockIdx.x * blockDim.x + threadIdx.x; // 512*5120
    if (i >= 512LL * 5120) return;
    long long r = i / 5120; int c = (int)(i % 5120);
    long long base = r * 1024;
    float v;
    if      (c < 1024) v = hf[base + c];
    else if (c < 2048) v = tf[base + c - 1024];
    else if (c < 3072) v = fabsf(hf[base + c - 2048] - tf[base + c - 2048]);
    else if (c < 4096) v = hf[base + c - 3072] * tf[base + c - 3072];
    else               v = pi[base + c - 4096];
    feat[i] = v;
}

// ---------------------------------------------------------------------------
// Host orchestration
// ---------------------------------------------------------------------------
static void launch_gemm(int nt, hipStream_t st,
                        const float* A, int lda, long long sA,
                        const int* aridx, int aridx_stride,
                        const float* Bm, int ldb, long long sB,
                        float* C, int ldc, long long sC,
                        const float* bias, int M, int N, int K, int batch, int flags)
{
    int tilesM = (M + 15) / 16, tilesN = (N + 63) / 64;
    int waves = tilesM * tilesN;
    dim3 g((waves + WPB - 1) / WPB, 1, batch), b(TPB);
    if (nt)
        gemm_kernel<1><<<g, b, 0, st>>>(A, lda, sA, Bm, ldb, sB, C, ldc, sC,
                                        bias, M, N, K, flags, aridx, aridx_stride);
    else
        gemm_kernel<0><<<g, b, 0, st>>>(A, lda, sA, Bm, ldb, sB, C, ldc, sC,
                                        bias, M, N, K, flags, aridx, aridx_stride);
}

extern "C" void kernel_launch(void* const* d_in, const int* in_sizes, int n_in,
                              void* d_out, int out_size, void* d_ws, size_t ws_size,
                              hipStream_t stream)
{
    // ---- inputs (setup_inputs dict order; params dict flattened in order) ----
    const int*   words = (const int*)  d_in[0];
    const int*   etype = (const int*)  d_in[1];
    const int*   eid   = (const int*)  d_in[2];
    const float* mask  = (const float*)d_in[3];
    const float* adj   = (const float*)d_in[4];
    const float* s2w   = (const float*)d_in[5];
    const int*   pairs = (const int*)  d_in[6];
    const int*   pidx  = (const int*)  d_in[7];
    const float* wemb = (const float*)d_in[8];
    const float* temb = (const float*)d_in[9];
    const float* iemb = (const float*)d_in[10];
    const float* WihF = (const float*)d_in[11];
    const float* WhhF = (const float*)d_in[12];
    const float* bF   = (const float*)d_in[13];
    const float* WihB = (const float*)d_in[14];
    const float* WhhB = (const float*)d_in[15];
    const float* bB   = (const float*)d_in[16];
    const float* g0W = (const float*)d_in[17]; const float* g0b = (const float*)d_in[18];
    const float* g1W = (const float*)d_in[19]; const float* g1b = (const float*)d_in[20];
    const float* g2W = (const float*)d_in[21]; const float* g2b = (const float*)d_in[22];
    const float* aW[2][4] = {
        { (const float*)d_in[23], (const float*)d_in[24], (const float*)d_in[25], (const float*)d_in[26] },
        { (const float*)d_in[27], (const float*)d_in[28], (const float*)d_in[29], (const float*)d_in[30] } };
    const float* r0Wih = (const float*)d_in[31];
    const float* r0Whh = (const float*)d_in[32];
    const float* r0b   = (const float*)d_in[33];
    const float* r1Wih = (const float*)d_in[34];
    const float* r1Whh = (const float*)d_in[35];
    const float* r1b   = (const float*)d_in[36];
    const float* paWin = (const float*)d_in[37];
    const float* paBin = (const float*)d_in[38];
    const float* paWout= (const float*)d_in[39];
    const float* paBout= (const float*)d_in[40];
    const float* predW = (const float*)d_in[41];
    const float* predB = (const float*)d_in[42];
    const float* outW  = (const float*)d_in[43];
    const float* outB  = (const float*)d_in[44];
    const float* binW  = (const float*)d_in[45];
    const float* binB  = (const float*)d_in[46];
    float* out = (float*)d_out;

    // ---- workspace bump allocator ----
    char* ws = (char*)d_ws; size_t off = 0;
    auto allocF = [&](long long nf) -> float* {
        float* p = (float*)(ws + off);
        off = (off + (size_t)nf * 4 + 255) & ~(size_t)255;
        return p;
    };
    float* emb   = allocF(8192LL * 140);
    float* GxF   = allocF(8192LL * 512);
    float* GxB   = allocF(8192LL * 512);
    float* feats = allocF(9216LL * 1024);   // [B*576, 1024]; cols: x0|g0|x2|x3
    float* Anorm = allocF(16LL * 576 * 576);
    float* tmpA  = allocF(9216LL * 256);
    float* tmpB  = allocF(9216LL * 256);
    float* qB    = allocF(9216LL * 256);
    float* kB    = allocF(9216LL * 256);
    float* vB    = allocF(9216LL * 256);
    float* attnO = allocF(9216LL * 256);
    float* scores= allocF(16LL * 576 * 576);
    float* hfeas = allocF(512LL * 1024);
    float* tfeas = allocF(512LL * 1024);
    float* G     = allocF(2048LL * 4096);
    float* hc    = allocF(4LL * 2048 * 1024);  // h0|c0|h1|c1
    float* h0 = hc, *c0 = hc + 2048LL*1024, *h1 = hc + 2*2048LL*1024, *c1 = hc + 3*2048LL*1024;
    float* pe    = allocF(2048LL * 1024);
    float* query = allocF(512LL * 1024);
    float* Qp    = allocF(512LL * 1024);
    float* Kp    = allocF(2048LL * 1024);
    float* Vp    = allocF(2048LL * 1024);
    float* oB    = allocF(512LL * 1024);
    float* pinfo = allocF(512LL * 1024);
    float* feat  = allocF(512LL * 5120);
    float* feat2 = allocF(512LL * 5120);
    int*   gidx  = (int*)allocF(16384);

    auto g1 = [](long long n) { return dim3((unsigned)((n + TPB - 1) / TPB)); };
    const long long FS = 576LL * 1024;   // features batch stride
    const long long TS = 576LL * 256;    // 256-wide temp batch stride
    const long long SS = 576LL * 576;    // scores batch stride

    // ---- 1. embeddings + encoder input GEMMs ----
    emb_kernel<<<g1(8192), TPB, 0, stream>>>(words, etype, eid, wemb, temb, iemb, emb);
    launch_gemm(1, stream, emb, 140, 0, nullptr, 0, WihF, 140, 0, GxF, 512, 0,
                nullptr, 8192, 512, 140, 1, 0);
    launch_gemm(1, stream, emb, 140, 0, nullptr, 0, WihB, 140, 0, GxB, 512, 0,
                nullptr, 8192, 512, 140, 1, 0);

    // ---- 2. fused BiLSTM (writes masked enc into features cols 0..255) ----
    enc_lstm_kernel<<<dim3(2), dim3(TPB), 0, stream>>>(GxF, GxB, WhhF, WhhB, bF, bB,
                                                       mask, feats);

    // ---- 3. ent_x = sub2words @ enc -> features rows 512..575, cols 0..255 ----
    launch_gemm(0, stream, s2w, 512, 64LL * 512, nullptr, 0,
                feats, 1024, FS, feats + 512LL * 1024, 1024, FS,
                nullptr, 64, 256, 512, 16, 0);

    // ---- 4. adjacency normalization ----
    adjnorm_kernel<<<g1(9216), TPB, 0, stream>>>(adj, Anorm);

    // gcn(colIn -> colOut): tmpA = Anorm @ h ; out = relu(tmpA @ W^T + b)
    auto gcn = [&](int colIn, const float* W, const float* b, float* Cout, int ldcO) {
        launch_gemm(0, stream, Anorm, 576, SS, nullptr, 0,
                    feats + colIn, 1024, FS, tmpA, 256, TS,
                    nullptr, 576, 256, 576, 16, 0);
        launch_gemm(1, stream, tmpA, 256, 0, nullptr, 0, W, 256, 0,
                    Cout, ldcO, 0, b, 9216, 256, 256, 1, 3 /*bias|relu*/);
    };
    // attn(i): q from features col qcol, k/v from tmpB(x1); result in attnO
    auto attn = [&](int ai, int qcol) {
        launch_gemm(1, stream, feats + qcol, 1024, 0, nullptr, 0, aW[ai][0], 256, 0,
                    qB, 256, 0, nullptr, 9216, 256, 256, 1, 0);
        launch_gemm(1, stream, tmpB, 256, 0, nullptr, 0, aW[ai][1], 256, 0,
                    kB, 256, 0, nullptr, 9216, 256, 256, 1, 0);
        launch_gemm(1, stream, tmpB, 256, 0, nullptr, 0, aW[ai][2], 256, 0,
                    vB, 256, 0, nullptr, 9216, 256, 256, 1, 0);
        launch_gemm(1, stream, qB, 256, TS, nullptr, 0, kB, 256, TS,
                    scores, 576, SS, nullptr, 576, 576, 256, 16, 0);
        softmax_rows_kernel<<<g1(9216), TPB, 0, stream>>>(scores, 9216, 576, 1.f / 16.f);
        launch_gemm(0, stream, scores, 576, SS, nullptr, 0, vB, 256, TS,
                    tmpA, 256, TS, nullptr, 576, 256, 576, 16, 0);
        launch_gemm(1, stream, tmpA, 256, 0, nullptr, 0, aW[ai][3], 256, 0,
                    attnO, 256, 0, nullptr, 9216, 256, 256, 1, 2 /*relu*/);
    };

    // ---- 5. GCN + attention stack (xs written into features column blocks) ----
    gcn(0,   g0W, g0b, feats + 256, 1024);                 // g0  -> cols 256..511
    gcn(256, g1W, g1b, tmpB, 256);                         // x1 (i=1)
    attn(0, 256);                                          // attn1(q=g0, kv=x1)
    add_into_features_kernel<<<g1(9216LL * 256), TPB, 0, stream>>>(feats, 512, tmpB, attnO);
    gcn(512, g2W, g2b, tmpB, 256);                         // x1 (i=2)
    attn(1, 512);                                          // attn2(q=x2, kv=x1)
    add_into_features_kernel<<<g1(9216LL * 256), TPB, 0, stream>>>(feats, 768, tmpB, attnO);

    // ---- 6. pair + path gathers ----
    pair_gather_kernel<<<g1(512LL * 1024), TPB, 0, stream>>>(feats, pairs, hfeas, tfeas);
    gidx_kernel<<<g1(16384), TPB, 0, stream>>>(pidx, gidx);

    // ---- 7. two-layer path LSTM, layers interleaved per timestep ----
    zero_kernel<<<g1(4LL * 2048 * 1024), TPB, 0, stream>>>(hc, 4LL * 2048 * 1024);
    for (int t = 0; t < 8; ++t) {
        // layer 0: G = features[gidx[:,t]] @ W0ih^T  (+= h0 @ W0hh^T)
        launch_gemm(1, stream, feats, 1024, 0, gidx + t, 8, r0Wih, 1024, 0,
                    G, 4096, 0, nullptr, 2048, 4096, 1024, 1, 0);
        launch_gemm(1, stream, h0, 1024, 0, nullptr, 0, r0Whh, 1024, 0,
                    G, 4096, 0, nullptr, 2048, 4096, 1024, 1, 4 /*accum*/);
        lstm_gate_kernel<<<g1(2048LL * 1024), TPB, 0, stream>>>(G, r0b, c0, h0,
                                                                nullptr, 0, 2048, 1024);
        // layer 1: G = h0 @ W1ih^T (+= h1 @ W1hh^T); pe = running max of h1
        launch_gemm(1, stream, h0, 1024, 0, nullptr, 0, r1Wih, 1024, 0,
                    G, 4096, 0, nullptr, 2048, 4096, 1024, 1, 0);
        launch_gemm(1, stream, h1, 1024, 0, nullptr, 0, r1Whh, 1024, 0,
                    G, 4096, 0, nullptr, 2048, 4096, 1024, 1, 4);
        lstm_gate_kernel<<<g1(2048LL * 1024), TPB, 0, stream>>>(G, r1b, c1, h1,
                                                                pe, t == 0, 2048, 1024);
    }

    // ---- 8. path attention ----
    sub_kernel<<<g1(512LL * 1024), TPB, 0, stream>>>(hfeas, tfeas, query, 512LL * 1024);
    launch_gemm(1, stream, query, 1024, 0, nullptr, 0, paWin, 1024, 0,
                Qp, 1024, 0, paBin, 512, 1024, 1024, 1, 1);
    launch_gemm(1, stream, pe, 1024, 0, nullptr, 0, paWin + 1024LL * 1024, 1024, 0,
                Kp, 1024, 0, paBin + 1024, 2048, 1024, 1024, 1, 1);
    launch_gemm(1, stream, pe, 1024, 0, nullptr, 0, paWin + 2048LL * 1024, 1024, 0,
                Vp, 1024, 0, paBin + 2048, 2048, 1024, 1024, 1, 1);
    pathattn_kernel<<<g1(2048), TPB, 0, stream>>>(Qp, Kp, Vp, oB);
    launch_gemm(1, stream, oB, 1024, 0, nullptr, 0, paWout, 1024, 0,
                pinfo, 1024, 0, paBout, 512, 1024, 1024, 1, 1);

    // ---- 9. heads ----
    featcat_kernel<<<g1(512LL * 5120), TPB, 0, stream>>>(hfeas, tfeas, pinfo, feat);
    launch_gemm(1, stream, feat, 5120, 0, nullptr, 0, predW, 5120, 0,
                feat2, 5120, 0, predB, 512, 5120, 5120, 1, 3 /*bias|relu*/);
    launch_gemm(1, stream, feat2, 5120, 0, nullptr, 0, outW, 5120, 0,
                out, 97, 0, outB, 512, 97, 5120, 1, 1);
    launch_gemm(1, stream, feat2, 5120, 0, nullptr, 0, binW, 5120, 0,
                out + 512LL * 97, 2, 0, binB, 512, 2, 5120, 1, 1);
    (void)in_sizes; (void)n_in; (void)out_size; (void)ws_size;
}
